// DINOv3StereoModel_4277787426960
// MI455X (gfx1250) — matrix operand order repaired and checked by reading, not compiled, not run on hardware
//
#include <hip/hip_runtime.h>
#include <hip/hip_bf16.h>
#include <math.h>

typedef __attribute__((ext_vector_type(16))) _Float16 v16h;
typedef __attribute__((ext_vector_type(8)))  float    v8f;

#define BB   2
#define CIN  768
#define HH   48
#define WWD  96
#define HWP  (HH*WWD)      /* 4608 */
#define H2D  24
#define W2D  48
#define P2D  (H2D*W2D)     /* 1152 */

// ---------------------------------------------------------------------------
// L2 normalize over channels (NCHW input) and pack to f16 NHWC (B,HW,C)
// ---------------------------------------------------------------------------
__global__ void k_l2norm_pack(const float* __restrict__ x, _Float16* __restrict__ y) {
  int n = blockIdx.x;                 // b*HW + p
  int b = n / HWP, p = n % HWP;
  __shared__ float red[256];
  float s = 0.f;
  for (int c = threadIdx.x; c < CIN; c += 256) {
    float v = x[((size_t)b * CIN + c) * HWP + p];
    s += v * v;
  }
  red[threadIdx.x] = s;
  __syncthreads();
  for (int st = 128; st > 0; st >>= 1) {
    if (threadIdx.x < st) red[threadIdx.x] += red[threadIdx.x + st];
    __syncthreads();
  }
  float inv = 1.f / fmaxf(sqrtf(red[0]), 1e-12f);
  for (int c = threadIdx.x; c < CIN; c += 256) {
    float v = x[((size_t)b * CIN + c) * HWP + p];
    y[(size_t)n * CIN + c] = (_Float16)(v * inv);
  }
}

// ---------------------------------------------------------------------------
// Register-blocked WMMA GEMM: 64x64 macro-tile per wave (4x4 WMMA tiles).
// C[M x N](f32) = A[M x K](f16 row major) * B[N x K](f16 row major)^T
// Requires M % 64 == 0, N % 64 == 0, K % 32 == 0.
// Double-buffered async global->LDS staging (ASYNCcnt): panel i+1 streams in
// while panel i feeds the matrix pipes (only latency hiding at 1 wave/SIMD).
// LDS rows padded to 40 halves (80 B) to spread banks.
// ---------------------------------------------------------------------------
#define LDSR 40
#define PANB (64 * LDSR * 2)   /* bytes per panel buffer: 5120 */
#define PANH (64 * LDSR)       /* halves per panel buffer */
__global__ __launch_bounds__(32, 1)
void k_gemm_wmma64(const _Float16* __restrict__ A,
                   const _Float16* __restrict__ Bm,
                   float* __restrict__ C, int M, int N, int K) {
  int nt = N >> 6;
  int tm = blockIdx.x / nt, tn = blockIdx.x % nt;
  int L  = threadIdx.x;
  __shared__ __align__(16) _Float16 As[2 * PANH];
  __shared__ __align__(16) _Float16 Bs[2 * PANH];
  v8f acc[16] = {};
  int m = L & 15, hi = L >> 4;
  const _Float16* apL0 = A  + (size_t)(tm * 64 + L)      * K;
  const _Float16* apL1 = A  + (size_t)(tm * 64 + L + 32) * K;
  const _Float16* bpL0 = Bm + (size_t)(tn * 64 + L)      * K;
  const _Float16* bpL1 = Bm + (size_t)(tn * 64 + L + 32) * K;
  // LDS byte addresses for this lane's destination rows (flat addr low 32 bits)
  unsigned la0 = (unsigned)(size_t)(As + L * LDSR);
  unsigned la1 = (unsigned)(size_t)(As + (L + 32) * LDSR);
  unsigned lb0 = (unsigned)(size_t)(Bs + L * LDSR);
  unsigned lb1 = (unsigned)(size_t)(Bs + (L + 32) * LDSR);
  // async copy 64x32 A panel + 64x32 B panel into buffer at byte offset bofs
  auto issue_panel = [&](unsigned bofs, int kk) {
#pragma unroll
    for (int q = 0; q < 4; ++q) {
      asm volatile("global_load_async_to_lds_b128 %0, %1, off"
                   :: "v"(la0 + bofs + q * 16), "v"((const void*)(apL0 + kk + q * 8))
                   : "memory");
      asm volatile("global_load_async_to_lds_b128 %0, %1, off"
                   :: "v"(la1 + bofs + q * 16), "v"((const void*)(apL1 + kk + q * 8))
                   : "memory");
      asm volatile("global_load_async_to_lds_b128 %0, %1, off"
                   :: "v"(lb0 + bofs + q * 16), "v"((const void*)(bpL0 + kk + q * 8))
                   : "memory");
      asm volatile("global_load_async_to_lds_b128 %0, %1, off"
                   :: "v"(lb1 + bofs + q * 16), "v"((const void*)(bpL1 + kk + q * 8))
                   : "memory");
    }
  };
  issue_panel(0u, 0);
  for (int k0 = 0; k0 < K; k0 += 32) {
    unsigned cur = ((k0 >> 5) & 1) ? (unsigned)PANB : 0u;   // byte offset of live buffer
    asm volatile("s_wait_asynccnt 0x0" ::: "memory");       // panel k0 resident
    __syncthreads();
    if (k0 + 32 < K) issue_panel(cur ^ (unsigned)PANB, k0 + 32);  // stream next panel
    if (k0 + 64 < K) {
      __builtin_prefetch(apL0 + k0 + 64, 0, 3);
      __builtin_prefetch(bpL0 + k0 + 64, 0, 3);
    }
    unsigned hofs = cur >> 1;                               // halves offset of live buffer
    v16h af[4];
#pragma unroll
    for (int ti = 0; ti < 4; ++ti) {
#pragma unroll
      for (int e = 0; e < 16; ++e) {
        int ka = (e & 7) + ((e >> 3) << 4) + hi * 8;        // A 16x32 f16 lane layout
        af[ti][e] = As[hofs + (ti * 16 + m) * LDSR + ka];
      }
    }
#pragma unroll
    for (int tj = 0; tj < 4; ++tj) {
      v16h bf;
#pragma unroll
      for (int e = 0; e < 16; ++e)
        bf[e] = Bs[hofs + (tj * 16 + m) * LDSR + (e & 7) + ((e >> 3) << 4) + hi * 8]; // B 32x16 f16 lane layout
#pragma unroll
      for (int ti = 0; ti < 4; ++ti)
        acc[ti * 4 + tj] = __builtin_amdgcn_wmma_f32_16x16x32_f16(
            false, af[ti], false, bf, (short)0, acc[ti * 4 + tj], false, false);
    }
  }
#pragma unroll
  for (int ti = 0; ti < 4; ++ti) {
#pragma unroll
    for (int tj = 0; tj < 4; ++tj) {
      int col = tn * 64 + tj * 16 + m;
#pragma unroll
      for (int v = 0; v < 8; ++v) {
        int row = tm * 64 + ti * 16 + v + hi * 8;
        C[(size_t)row * N + col] = acc[ti * 4 + tj][v];
      }
    }
  }
}

// ---------------------------------------------------------------------------
// Fallback WMMA GEMM: one 16x16 tile per wave (used when N % 64 != 0).
// Synchronous LDS staging (kept as the vanilla path).
// ---------------------------------------------------------------------------
__global__ __launch_bounds__(32, 1)
void k_gemm_wmma(const _Float16* __restrict__ A,
                 const _Float16* __restrict__ Bm,
                 float* __restrict__ C, int M, int N, int K) {
  int nt = N >> 4;
  int tm = blockIdx.x / nt, tn = blockIdx.x % nt;
  int L  = threadIdx.x;
  __shared__ __align__(16) _Float16 As[16 * 32];
  __shared__ __align__(16) _Float16 Bs[16 * 32];
  v8f acc = {};
  int r = L >> 1, c0 = (L & 1) * 16;
  const _Float16* ap = A  + (size_t)(tm * 16 + r) * K + c0;
  const _Float16* bp = Bm + (size_t)(tn * 16 + r) * K + c0;
  int m = L & 15, hi = L >> 4;
  for (int k0 = 0; k0 < K; k0 += 32) {
    const uint4* pa = (const uint4*)(ap + k0);
    const uint4* pb = (const uint4*)(bp + k0);
    uint4 a0 = pa[0], a1 = pa[1];
    uint4 b0 = pb[0], b1 = pb[1];
    uint4* da = (uint4*)(As + r * 32 + c0);
    uint4* db = (uint4*)(Bs + r * 32 + c0);
    da[0] = a0; da[1] = a1;
    db[0] = b0; db[1] = b1;
    __syncthreads();
    if (k0 + 32 < K) {
      __builtin_prefetch(ap + k0 + 32, 0, 3);
      __builtin_prefetch(bp + k0 + 32, 0, 3);
    }
    v16h av, bv;
#pragma unroll
    for (int e = 0; e < 16; ++e) {
      int ka = (e & 7) + ((e >> 3) << 4) + hi * 8;
      av[e] = As[m * 32 + ka];
      bv[e] = Bs[m * 32 + ka];
    }
    acc = __builtin_amdgcn_wmma_f32_16x16x32_f16(false, av, false, bv,
                                                 (short)0, acc, false, false);
    __syncthreads();
  }
#pragma unroll
  for (int v = 0; v < 8; ++v) {
    int row = tm * 16 + v + hi * 8;
    int col = tn * 16 + m;
    C[(size_t)row * N + col] = acc[v];
  }
}

// ---------------------------------------------------------------------------
// 2x2 average pool over per-image (N images of Hs x Ws)
// ---------------------------------------------------------------------------
__global__ void k_avgpool2(const float* __restrict__ src, float* __restrict__ dst,
                           int Nimg, int Hs, int Ws) {
  int Hd = Hs >> 1, Wd = Ws >> 1;
  size_t total = (size_t)Nimg * Hd * Wd;
  size_t i = (size_t)blockIdx.x * blockDim.x + threadIdx.x;
  if (i >= total) return;
  int wd = (int)(i % Wd); size_t t = i / Wd;
  int hd = (int)(t % Hd); size_t n = t / Hd;
  const float* im = src + n * (size_t)Hs * Ws;
  float v = im[(2 * hd) * Ws + 2 * wd] + im[(2 * hd) * Ws + 2 * wd + 1] +
            im[(2 * hd + 1) * Ws + 2 * wd] + im[(2 * hd + 1) * Ws + 2 * wd + 1];
  dst[i] = v * 0.25f;
}

// Pack f32 weights (O x K contiguous) to f16 (O x Kp), zero pad K..Kp
__global__ void k_pack_w(const float* __restrict__ src, _Float16* __restrict__ dst,
                         int O, int K, int Kp) {
  size_t i = (size_t)blockIdx.x * blockDim.x + threadIdx.x;
  if (i >= (size_t)O * Kp) return;
  int k = (int)(i % Kp); int o = (int)(i / Kp);
  dst[i] = (k < K) ? (_Float16)src[(size_t)o * K + k] : (_Float16)0.f;
}

__global__ void k_cast_h(const float* __restrict__ src, _Float16* __restrict__ dst, size_t n) {
  size_t i = (size_t)blockIdx.x * blockDim.x + threadIdx.x;
  if (i < n) dst[i] = (_Float16)src[i];
}

// im2col 3x3 pad1, stride 1 or 2, NHWC input (Bn,Hh*Ww,Cc), K order = c*9+ky*3+kx,
// zero-padded to Kp columns, f16 output.
__global__ void k_im2col(const float* __restrict__ x, _Float16* __restrict__ out,
                         int Bn, int Hh, int Ww, int Cc, int Kp, int stride) {
  int Ho = Hh / stride, Wo = Ww / stride;
  size_t total = (size_t)Bn * Ho * Wo * Kp;
  size_t i = (size_t)blockIdx.x * blockDim.x + threadIdx.x;
  if (i >= total) return;
  int k = (int)(i % Kp); size_t rowI = i / Kp;
  int K = Cc * 9;
  _Float16 val = (_Float16)0.f;
  if (k < K) {
    int c = k / 9, r9 = k % 9, ky = r9 / 3, kx = r9 % 3;
    int p = (int)(rowI % (Ho * Wo)); int b = (int)(rowI / (Ho * Wo));
    int h = p / Wo, w = p % Wo;
    int ih = h * stride + ky - 1, iw = w * stride + kx - 1;
    if (ih >= 0 && ih < Hh && iw >= 0 && iw < Ww)
      val = (_Float16)x[((size_t)b * Hh * Ww + (size_t)ih * Ww + iw) * Cc + c];
  }
  out[i] = val;
}

// y[row,n] = act(y[row,n] + bias[n]) ; mode 0 none, 1 relu, 2 sigmoid, 3 tanh
__global__ void k_bias_act(float* __restrict__ y, const float* __restrict__ b,
                           size_t rows, int N, int mode) {
  size_t total = rows * N;
  size_t i = (size_t)blockIdx.x * blockDim.x + threadIdx.x;
  if (i >= total) return;
  float v = y[i] + b[i % N];
  if (mode == 1) v = fmaxf(v, 0.f);
  else if (mode == 2) v = 1.f / (1.f + expf(-v));
  else if (mode == 3) v = tanhf(v);
  y[i] = v;
}

// GroupNorm stats over (HW x 16ch) per (b, group); x is (B,HW,128)
__global__ void k_gn_stats(const float* __restrict__ x, float* __restrict__ st) {
  int b = blockIdx.x / 8, g = blockIdx.x % 8;
  __shared__ float s1[256], s2[256];
  float a = 0.f, q = 0.f;
  for (int i = threadIdx.x; i < HWP * 16; i += 256) {
    int p = i >> 4, c = i & 15;
    float v = x[((size_t)b * HWP + p) * 128 + g * 16 + c];
    a += v; q += v * v;
  }
  s1[threadIdx.x] = a; s2[threadIdx.x] = q;
  __syncthreads();
  for (int stp = 128; stp > 0; stp >>= 1) {
    if (threadIdx.x < stp) { s1[threadIdx.x] += s1[threadIdx.x + stp];
                             s2[threadIdx.x] += s2[threadIdx.x + stp]; }
    __syncthreads();
  }
  if (threadIdx.x == 0) {
    float cnt = (float)(HWP * 16);
    float mu = s1[0] / cnt;
    float var = s2[0] / cnt - mu * mu;
    st[(b * 8 + g) * 2]     = mu;
    st[(b * 8 + g) * 2 + 1] = var;
  }
}

__global__ void k_gn_apply(const float* __restrict__ x, float* __restrict__ y,
                           const float* __restrict__ st, const float* __restrict__ gam,
                           const float* __restrict__ bet, const float* __restrict__ resid,
                           int relu) {
  size_t total = (size_t)BB * HWP * 128;
  size_t i = (size_t)blockIdx.x * blockDim.x + threadIdx.x;
  if (i >= total) return;
  int c = (int)(i % 128); size_t n = i / 128; int b = (int)(n / HWP);
  int g = c >> 4;
  float mu = st[(b * 8 + g) * 2], var = st[(b * 8 + g) * 2 + 1];
  float v = (x[i] - mu) * rsqrtf(var + 1e-5f) * gam[c] + bet[c];
  if (resid) v += resid[i];
  if (relu) v = fmaxf(v, 0.f);
  y[i] = v;
}

// split cnet (B,HW,256) -> net = tanh(ch 0..127), inp = relu(ch 128..255)
__global__ void k_split_cnet(const float* __restrict__ cn, float* __restrict__ net,
                             float* __restrict__ inp) {
  size_t total = (size_t)BB * HWP * 128;
  size_t i = (size_t)blockIdx.x * blockDim.x + threadIdx.x;
  if (i >= total) return;
  int c = (int)(i % 128); size_t n = i / 128;
  net[i] = tanhf(cn[n * 256 + c]);
  inp[i] = fmaxf(cn[n * 256 + 128 + c], 0.f);
}

// Corr pyramid lookup; replicates reference grid-sample quirks exactly.
__global__ void k_corr_lookup(const float* __restrict__ p0, const float* __restrict__ p1,
                              const float* __restrict__ p2, const float* __restrict__ p3,
                              const float* __restrict__ flow, float* __restrict__ out) {
  int n = blockIdx.x;                      // b*HW + p
  int p = n % HWP; int h = p / WWD, w = p % WWD;
  float fx = flow[(size_t)n * 2], fy = flow[(size_t)n * 2 + 1];
  float cx = (float)w - fx, cy = (float)h - fy;
  const float* pyr[4] = {p0, p1, p2, p3};
  for (int ch = threadIdx.x; ch < 324; ch += blockDim.x) {
    int l = ch / 81, d = ch % 81;
    float da = (float)(d / 9) - 4.f;       // delta[...,0] -> added to x (torch quirk)
    float db = (float)(d % 9) - 4.f;       // delta[...,1] -> added to y
    int Hl = HH >> l, Wl = WWD >> l;
    float inv = 1.f / (float)(1 << l);
    float gx = cx * inv + da, gy = cy * inv + db;
    float xx = (gx + 1.f) * 0.5f * (float)(Wl - 1);
    float yy = (gy + 1.f) * 0.5f * (float)(Hl - 1);
    xx = fminf(fmaxf(xx, 0.f), (float)(Wl - 1));
    yy = fminf(fmaxf(yy, 0.f), (float)(Hl - 1));
    int x0 = (int)floorf(xx), y0 = (int)floorf(yy);
    float wx = xx - (float)x0, wy = yy - (float)y0;
    int x1 = min(x0 + 1, Wl - 1), y1 = min(y0 + 1, Hl - 1);
    const float* im = pyr[l] + (size_t)n * Hl * Wl;
    float v = im[y0 * Wl + x0] * (1 - wx) * (1 - wy) + im[y0 * Wl + x1] * wx * (1 - wy)
            + im[y1 * Wl + x0] * (1 - wx) * wy       + im[y1 * Wl + x1] * wx * wy;
    out[(size_t)n * 324 + ch] = v;
  }
}

__global__ void k_concat2(const float* __restrict__ a, int ca,
                          const float* __restrict__ b, int cb,
                          float* __restrict__ o, size_t rows) {
  int ct = ca + cb;
  size_t total = rows * ct;
  size_t i = (size_t)blockIdx.x * blockDim.x + threadIdx.x;
  if (i >= total) return;
  int c = (int)(i % ct); size_t n = i / ct;
  o[i] = (c < ca) ? a[n * ca + c] : b[n * cb + (c - ca)];
}

// bilinear x2 upsample (align_corners=False), src (B,P2,32) -> dst (B,HW,32)
__global__ void k_upsample2(const float* __restrict__ src, float* __restrict__ dst) {
  size_t total = (size_t)BB * HWP * 32;
  size_t i = (size_t)blockIdx.x * blockDim.x + threadIdx.x;
  if (i >= total) return;
  int c = (int)(i % 32); size_t t = i / 32;
  int p = (int)(t % HWP); int b = (int)(t / HWP);
  int h = p / WWD, w = p % WWD;
  float sx = (w + 0.5f) * 0.5f - 0.5f, sy = (h + 0.5f) * 0.5f - 0.5f;
  sx = fminf(fmaxf(sx, 0.f), (float)(W2D - 1));
  sy = fminf(fmaxf(sy, 0.f), (float)(H2D - 1));
  int x0 = (int)floorf(sx), y0 = (int)floorf(sy);
  float wx = sx - (float)x0, wy = sy - (float)y0;
  int x1 = min(x0 + 1, W2D - 1), y1 = min(y0 + 1, H2D - 1);
  const float* base = src + (size_t)b * P2D * 32;
  float v00 = base[((size_t)y0 * W2D + x0) * 32 + c];
  float v01 = base[((size_t)y0 * W2D + x1) * 32 + c];
  float v10 = base[((size_t)y1 * W2D + x0) * 32 + c];
  float v11 = base[((size_t)y1 * W2D + x1) * 32 + c];
  dst[i] = v00 * (1 - wx) * (1 - wy) + v01 * wx * (1 - wy)
         + v10 * (1 - wx) * wy + v11 * wx * wy;
}

__global__ void k_mul(const float* __restrict__ a, const float* __restrict__ b,
                      float* __restrict__ o, size_t n) {
  size_t i = (size_t)blockIdx.x * blockDim.x + threadIdx.x;
  if (i < n) o[i] = a[i] * b[i];
}

__global__ void k_gru(float* __restrict__ net, const float* __restrict__ z,
                      const float* __restrict__ q, size_t n) {
  size_t i = (size_t)blockIdx.x * blockDim.x + threadIdx.x;
  if (i < n) net[i] = (1.f - z[i]) * net[i] + z[i] * q[i];
}

__global__ void k_fill(float* __restrict__ p, size_t n, float v) {
  size_t i = (size_t)blockIdx.x * blockDim.x + threadIdx.x;
  if (i < n) p[i] = v;
}

// flow head: dlt = tanh(dot(dh,w2)+b2); flow += dlt (both ch); write pred NCHW
__global__ void k_h2_flow(const float* __restrict__ dh, const float* __restrict__ w2,
                          const float* __restrict__ b2, float* __restrict__ flow,
                          float* __restrict__ outp) {
  int n = blockIdx.x * blockDim.x + threadIdx.x;
  if (n >= BB * HWP) return;
  float s = 0.f;
  for (int c = 0; c < 256; ++c) s += dh[(size_t)n * 256 + c] * w2[c];
  float dlt = tanhf(s + b2[0]);     // SCALE == 1.0
  float fx = flow[(size_t)n * 2]     + dlt;
  float fy = flow[(size_t)n * 2 + 1] + dlt;
  flow[(size_t)n * 2]     = fx;
  flow[(size_t)n * 2 + 1] = fy;
  int b = n / HWP, p = n % HWP;
  outp[((size_t)b * 2 + 0) * HWP + p] = fx;
  outp[((size_t)b * 2 + 1) * HWP + p] = fy;
}

// ---------------------------------------------------------------------------
extern "C" void kernel_launch(void* const* d_in, const int* in_sizes, int n_in,
                              void* d_out, int out_size, void* d_ws, size_t ws_size,
                              hipStream_t stream) {
  (void)in_sizes; (void)n_in; (void)out_size; (void)ws_size;
  const float* fmap1 = (const float*)d_in[0];
  const float* fmap2 = (const float*)d_in[1];
  // params flattened in sorted-key order (insertion order of _init_params)
  const float* cin_b  = (const float*)d_in[2];
  const float* cin_w  = (const float*)d_in[3];
  const float* cout_b = (const float*)d_in[4];
  const float* cout_w = (const float*)d_in[5];
  const float* gq_b   = (const float*)d_in[6];
  const float* gq_w   = (const float*)d_in[7];
  const float* gr_b   = (const float*)d_in[8];
  const float* gr_w   = (const float*)d_in[9];
  const float* gz_b   = (const float*)d_in[10];
  const float* gz_w   = (const float*)d_in[11];
  const float* h1_b   = (const float*)d_in[12];
  const float* h1_w   = (const float*)d_in[13];
  const float* h2_b   = (const float*)d_in[14];
  const float* h2_w   = (const float*)d_in[15];
  const float* me1_b  = (const float*)d_in[16];
  const float* me1_w  = (const float*)d_in[17];
  const float* me2_b  = (const float*)d_in[18];
  const float* me2_w  = (const float*)d_in[19];
  const float* me3_b  = (const float*)d_in[20];
  const float* me3_w  = (const float*)d_in[21];
  const float* r1c1_w = (const float*)d_in[22];
  const float* r1c2_w = (const float*)d_in[23];
  const float* r1g1_b = (const float*)d_in[24];
  const float* r1g1_g = (const float*)d_in[25];
  const float* r1g2_b = (const float*)d_in[26];
  const float* r1g2_g = (const float*)d_in[27];
  const float* r2c1_w = (const float*)d_in[28];
  const float* r2c2_w = (const float*)d_in[29];
  const float* r2g1_b = (const float*)d_in[30];
  const float* r2g1_g = (const float*)d_in[31];
  const float* r2g2_b = (const float*)d_in[32];
  const float* r2g2_g = (const float*)d_in[33];

  char* ws = (char*)d_ws;
  size_t off = 0;
  auto alloc = [&](size_t bytes) -> char* {
    char* p = ws + off;
    off += (bytes + 255) & ~(size_t)255;
    return p;
  };
  const size_t M  = (size_t)BB * HWP;   // 9216
  const size_t Mh = (size_t)BB * P2D;   // 2304

  _Float16* f1h  = (_Float16*)alloc(M * CIN * 2);
  _Float16* f2h  = (_Float16*)alloc(M * CIN * 2);
  float* pyr0 = (float*)alloc((size_t)BB * HWP * HWP * 4);
  float* pyr1 = (float*)alloc((size_t)BB * HWP * (HWP / 4) * 4);
  float* pyr2 = (float*)alloc((size_t)BB * HWP * (HWP / 16) * 4);
  float* pyr3 = (float*)alloc((size_t)BB * HWP * (HWP / 64) * 4);
  _Float16* imb = (_Float16*)alloc(M * 2592 * 2);
  _Float16* wp_cin  = (_Float16*)alloc((size_t)128 * 768 * 2);
  _Float16* wp_cout = (_Float16*)alloc((size_t)256 * 128 * 2);
  _Float16* wp_gq   = (_Float16*)alloc((size_t)128 * 2592 * 2);
  _Float16* wp_gr   = (_Float16*)alloc((size_t)128 * 2592 * 2);
  _Float16* wp_gz   = (_Float16*)alloc((size_t)128 * 2592 * 2);
  _Float16* wp_h1   = (_Float16*)alloc((size_t)256 * 1152 * 2);
  _Float16* wp_me1  = (_Float16*)alloc((size_t)128 * 2944 * 2);
  _Float16* wp_me2  = (_Float16*)alloc((size_t)64 * 1152 * 2);
  _Float16* wp_me3  = (_Float16*)alloc((size_t)32 * 576 * 2);
  _Float16* wp_r1c1 = (_Float16*)alloc((size_t)128 * 1152 * 2);
  _Float16* wp_r1c2 = (_Float16*)alloc((size_t)128 * 1152 * 2);
  _Float16* wp_r2c1 = (_Float16*)alloc((size_t)128 * 1152 * 2);
  _Float16* wp_r2c2 = (_Float16*)alloc((size_t)128 * 1152 * 2);
  float* xbuf  = (float*)alloc(M * 128 * 4);
  float* tbuf  = (float*)alloc(M * 128 * 4);
  float* gconv = (float*)alloc(M * 128 * 4);
  float* cnetb = (float*)alloc(M * 256 * 4);
  float* netb  = (float*)alloc(M * 128 * 4);
  float* inpb  = (float*)alloc(M * 128 * 4);
  float* flowb = (float*)alloc(M * 2 * 4);
  float* corrb = (float*)alloc(M * 324 * 4);
  float* mcat  = (float*)alloc(M * 326 * 4);
  float* m1    = (float*)alloc(Mh * 128 * 4);
  float* m2    = (float*)alloc(Mh * 64 * 4);
  float* m3    = (float*)alloc(Mh * 32 * 4);
  float* mup   = (float*)alloc(M * 32 * 4);
  float* inpcat= (float*)alloc(M * 160 * 4);
  float* hxb   = (float*)alloc(M * 288 * 4);
  float* zb    = (float*)alloc(M * 128 * 4);
  float* rb    = (float*)alloc(M * 128 * 4);
  float* qb    = (float*)alloc(M * 128 * 4);
  float* rnb   = (float*)alloc(M * 128 * 4);
  float* dhb   = (float*)alloc(M * 256 * 4);
  float* gnst  = (float*)alloc(256);

  auto g1 = [](size_t n) { return dim3((unsigned)((n + 255) / 256)); };
  auto gemm = [&](const _Float16* A, const _Float16* B, float* C, int Mm, int Nn, int Kk) {
    if ((Mm % 64) == 0 && (Nn % 64) == 0) {
      k_gemm_wmma64<<<dim3((unsigned)((Mm / 64) * (Nn / 64))), dim3(32), 0, stream>>>(A, B, C, Mm, Nn, Kk);
    } else {
      k_gemm_wmma<<<dim3((unsigned)((Mm / 16) * (Nn / 16))), dim3(32), 0, stream>>>(A, B, C, Mm, Nn, Kk);
    }
  };

  // ---- weight packing (f32 -> f16, K zero-padded) ----
  k_pack_w<<<g1((size_t)128*768), 256, 0, stream>>>(cin_w, wp_cin, 128, 768, 768);
  k_pack_w<<<g1((size_t)256*128), 256, 0, stream>>>(cout_w, wp_cout, 256, 128, 128);
  k_pack_w<<<g1((size_t)128*2592), 256, 0, stream>>>(gq_w, wp_gq, 128, 2592, 2592);
  k_pack_w<<<g1((size_t)128*2592), 256, 0, stream>>>(gr_w, wp_gr, 128, 2592, 2592);
  k_pack_w<<<g1((size_t)128*2592), 256, 0, stream>>>(gz_w, wp_gz, 128, 2592, 2592);
  k_pack_w<<<g1((size_t)256*1152), 256, 0, stream>>>(h1_w, wp_h1, 256, 1152, 1152);
  k_pack_w<<<g1((size_t)128*2944), 256, 0, stream>>>(me1_w, wp_me1, 128, 2934, 2944);
  k_pack_w<<<g1((size_t)64*1152), 256, 0, stream>>>(me2_w, wp_me2, 64, 1152, 1152);
  k_pack_w<<<g1((size_t)32*576), 256, 0, stream>>>(me3_w, wp_me3, 32, 576, 576);
  k_pack_w<<<g1((size_t)128*1152), 256, 0, stream>>>(r1c1_w, wp_r1c1, 128, 1152, 1152);
  k_pack_w<<<g1((size_t)128*1152), 256, 0, stream>>>(r1c2_w, wp_r1c2, 128, 1152, 1152);
  k_pack_w<<<g1((size_t)128*1152), 256, 0, stream>>>(r2c1_w, wp_r2c1, 128, 1152, 1152);
  k_pack_w<<<g1((size_t)128*1152), 256, 0, stream>>>(r2c2_w, wp_r2c2, 128, 1152, 1152);

  // ---- normalize + pack feature maps ----
  k_l2norm_pack<<<dim3((unsigned)M), 256, 0, stream>>>(fmap1, f1h);
  k_l2norm_pack<<<dim3((unsigned)M), 256, 0, stream>>>(fmap2, f2h);

  // ---- correlation volume (per batch): rows = f2 pixel, cols = f1 pixel ----
  for (int b = 0; b < BB; ++b) {
    gemm(f2h + (size_t)b * HWP * CIN, f1h + (size_t)b * HWP * CIN,
         pyr0 + (size_t)b * HWP * HWP, HWP, HWP, CIN);
  }
  // ---- pyramid ----
  k_avgpool2<<<g1((size_t)BB*HWP*(HWP/4)), 256, 0, stream>>>(pyr0, pyr1, BB*HWP, HH, WWD);
  k_avgpool2<<<g1((size_t)BB*HWP*(HWP/16)), 256, 0, stream>>>(pyr1, pyr2, BB*HWP, HH/2, WWD/2);
  k_avgpool2<<<g1((size_t)BB*HWP*(HWP/64)), 256, 0, stream>>>(pyr2, pyr3, BB*HWP, HH/4, WWD/4);

  // ---- context network ----
  gemm(f1h, wp_cin, xbuf, (int)M, 128, 768);
  k_bias_act<<<g1(M*128), 256, 0, stream>>>(xbuf, cin_b, M, 128, 1);
  // resblock 1
  k_im2col<<<g1(M*1152), 256, 0, stream>>>(xbuf, imb, BB, HH, WWD, 128, 1152, 1);
  gemm(imb, wp_r1c1, gconv, (int)M, 128, 1152);
  k_gn_stats<<<dim3(BB*8), 256, 0, stream>>>(gconv, gnst);
  k_gn_apply<<<g1(M*128), 256, 0, stream>>>(gconv, tbuf, gnst, r1g1_g, r1g1_b, nullptr, 1);
  k_im2col<<<g1(M*1152), 256, 0, stream>>>(tbuf, imb, BB, HH, WWD, 128, 1152, 1);
  gemm(imb, wp_r1c2, gconv, (int)M, 128, 1152);
  k_gn_stats<<<dim3(BB*8), 256, 0, stream>>>(gconv, gnst);
  k_gn_apply<<<g1(M*128), 256, 0, stream>>>(gconv, xbuf, gnst, r1g2_g, r1g2_b, xbuf, 1);
  // resblock 2
  k_im2col<<<g1(M*1152), 256, 0, stream>>>(xbuf, imb, BB, HH, WWD, 128, 1152, 1);
  gemm(imb, wp_r2c1, gconv, (int)M, 128, 1152);
  k_gn_stats<<<dim3(BB*8), 256, 0, stream>>>(gconv, gnst);
  k_gn_apply<<<g1(M*128), 256, 0, stream>>>(gconv, tbuf, gnst, r2g1_g, r2g1_b, nullptr, 1);
  k_im2col<<<g1(M*1152), 256, 0, stream>>>(tbuf, imb, BB, HH, WWD, 128, 1152, 1);
  gemm(imb, wp_r2c2, gconv, (int)M, 128, 1152);
  k_gn_stats<<<dim3(BB*8), 256, 0, stream>>>(gconv, gnst);
  k_gn_apply<<<g1(M*128), 256, 0, stream>>>(gconv, xbuf, gnst, r2g2_g, r2g2_b, xbuf, 1);
  // cout 1x1 -> split into net/inp
  k_cast_h<<<g1(M*128), 256, 0, stream>>>(xbuf, imb, M * 128);
  gemm(imb, wp_cout, cnetb, (int)M, 256, 128);
  k_bias_act<<<g1(M*256), 256, 0, stream>>>(cnetb, cout_b, M, 256, 0);
  k_split_cnet<<<g1(M*128), 256, 0, stream>>>(cnetb, netb, inpb);
  // flow = 0
  k_fill<<<g1(M*2), 256, 0, stream>>>(flowb, M * 2, 0.f);

  float* outp = (float*)d_out;
  for (int t = 0; t < 4; ++t) {
    // corr lookup
    k_corr_lookup<<<dim3((unsigned)M), 128, 0, stream>>>(pyr0, pyr1, pyr2, pyr3, flowb, corrb);
    // motion encoder
    k_concat2<<<g1(M*326), 256, 0, stream>>>(flowb, 2, corrb, 324, mcat, M);
    k_im2col<<<g1(Mh*2944), 256, 0, stream>>>(mcat, imb, BB, HH, WWD, 326, 2944, 2);
    gemm(imb, wp_me1, m1, (int)Mh, 128, 2944);
    k_bias_act<<<g1(Mh*128), 256, 0, stream>>>(m1, me1_b, Mh, 128, 1);
    k_im2col<<<g1(Mh*1152), 256, 0, stream>>>(m1, imb, BB, H2D, W2D, 128, 1152, 1);
    gemm(imb, wp_me2, m2, (int)Mh, 64, 1152);
    k_bias_act<<<g1(Mh*64), 256, 0, stream>>>(m2, me2_b, Mh, 64, 1);
    k_im2col<<<g1(Mh*576), 256, 0, stream>>>(m2, imb, BB, H2D, W2D, 64, 576, 1);
    gemm(imb, wp_me3, m3, (int)Mh, 32, 576);
    k_bias_act<<<g1(Mh*32), 256, 0, stream>>>(m3, me3_b, Mh, 32, 1);
    k_upsample2<<<g1(M*32), 256, 0, stream>>>(m3, mup);
    // GRU
    k_concat2<<<g1(M*160), 256, 0, stream>>>(inpb, 128, mup, 32, inpcat, M);
    k_concat2<<<g1(M*288), 256, 0, stream>>>(netb, 128, inpcat, 160, hxb, M);
    k_im2col<<<g1(M*2592), 256, 0, stream>>>(hxb, imb, BB, HH, WWD, 288, 2592, 1);
    gemm(imb, wp_gz, zb, (int)M, 128, 2592);
    k_bias_act<<<g1(M*128), 256, 0, stream>>>(zb, gz_b, M, 128, 2);
    gemm(imb, wp_gr, rb, (int)M, 128, 2592);
    k_bias_act<<<g1(M*128), 256, 0, stream>>>(rb, gr_b, M, 128, 2);
    k_mul<<<g1(M*128), 256, 0, stream>>>(rb, netb, rnb, M * 128);
    k_concat2<<<g1(M*288), 256, 0, stream>>>(rnb, 128, inpcat, 160, hxb, M);
    k_im2col<<<g1(M*2592), 256, 0, stream>>>(hxb, imb, BB, HH, WWD, 288, 2592, 1);
    gemm(imb, wp_gq, qb, (int)M, 128, 2592);
    k_bias_act<<<g1(M*128), 256, 0, stream>>>(qb, gq_b, M, 128, 3);
    k_gru<<<g1(M*128), 256, 0, stream>>>(netb, zb, qb, M * 128);
    // flow head
    k_im2col<<<g1(M*1152), 256, 0, stream>>>(netb, imb, BB, HH, WWD, 128, 1152, 1);
    gemm(imb, wp_h1, dhb, (int)M, 256, 1152);
    k_bias_act<<<g1(M*256), 256, 0, stream>>>(dhb, h1_b, M, 256, 1);
    k_h2_flow<<<g1(M), 256, 0, stream>>>(dhb, h2_w, h2_b, flowb, outp + (size_t)t * BB * 2 * HWP);
  }
}
